// MentionScore_13091060318541
// MI455X (gfx1250) — matrix operand order repaired and checked by reading, not compiled, not run on hardware
//
#include <hip/hip_runtime.h>
#include <hip/hip_bf16.h>

// Problem constants (from reference)
#define T_TOK   4096
#define D_FEAT  512
#define L_SPAN  10
#define H_DIM   512
#define N_SPAN  40960
#define G_DIM   1536          // 3*D
#define K_PRUNE 1638          // int(0.4 * 4096)

typedef __attribute__((ext_vector_type(16))) __bf16 v16bf;
typedef __attribute__((ext_vector_type(8)))  float  v8f;

union Frag {
    unsigned int u[8];
    uint4        q[2];
    v16bf        v;
};

__device__ __forceinline__ unsigned short f2bf(float f) {
    unsigned int x = __builtin_bit_cast(unsigned int, f);
    return (unsigned short)((x + 0x7FFFu + ((x >> 16) & 1u)) >> 16);  // RNE
}

#if defined(__has_builtin)
#if __has_builtin(__builtin_amdgcn_cvt_pk_bf16_f32)
#define HAVE_CVT_PK_BF16 1
#endif
#endif

#ifdef HAVE_CVT_PK_BF16
typedef __attribute__((ext_vector_type(2))) __bf16 v2bf;
__device__ __forceinline__ unsigned int pack2bf(float lo, float hi) {
    v2bf r = __builtin_amdgcn_cvt_pk_bf16_f32(lo, hi);
    return __builtin_bit_cast(unsigned int, r);
}
#else
__device__ __forceinline__ unsigned int pack2bf(float lo, float hi) {
    return (unsigned int)f2bf(lo) | ((unsigned int)f2bf(hi) << 16);
}
#endif

__device__ __forceinline__ float bf2f(unsigned short u) {
    return __builtin_bit_cast(float, ((unsigned int)u) << 16);
}
__device__ __forceinline__ float wave_sum(float v) {
    for (int o = 16; o > 0; o >>= 1) v += __shfl_xor(v, o, 32);
    return v;
}
__device__ __forceinline__ float wave_max(float v) {
    for (int o = 16; o > 0; o >>= 1) v = fmaxf(v, __shfl_xor(v, o, 32));
    return v;
}

// ---------------------------------------------------------------------------
// Transpose + convert fp32 weight [K,N] -> bf16 Wt [N,K]
// ---------------------------------------------------------------------------
__global__ void convert_wt(const float* __restrict__ W, unsigned short* __restrict__ Wt,
                           int K, int N) {
    int idx = blockIdx.x * 256 + threadIdx.x;
    if (idx >= K * N) return;
    int k = idx / N, n = idx % N;
    Wt[(size_t)n * K + k] = f2bf(W[(size_t)k * N + n]);
}

// Elementwise fp32 -> bf16, 8 elements per thread (vectorized)
__global__ void convert_vec(const float* __restrict__ X, unsigned short* __restrict__ Y, int n8) {
    int i = blockIdx.x * 256 + threadIdx.x;
    if (i >= n8) return;
    const float4* xp = (const float4*)(X + (size_t)i * 8);
    float4 a = xp[0], b = xp[1];
    union { unsigned int u[4]; uint4 q; } p;
    p.u[0] = pack2bf(a.x, a.y); p.u[1] = pack2bf(a.z, a.w);
    p.u[2] = pack2bf(b.x, b.y); p.u[3] = pack2bf(b.z, b.w);
    *(uint4*)(Y + (size_t)i * 8) = p.q;
}

// ---------------------------------------------------------------------------
// Fragment loader (all bf16). aoff folds arow*K + 16*half; boff folds
// (n0+l16)*K + 16*half. K offsets multiples of 8 elements -> 16B aligned.
// ---------------------------------------------------------------------------
__device__ __forceinline__ void load_frags(const unsigned short* __restrict__ A,
                                           const unsigned short* __restrict__ Bt,
                                           size_t aoff, size_t boff, size_t Kst,
                                           int k0, Frag& a, Frag* b) {
    const unsigned short* ap = A + aoff + k0;
    a.q[0] = *(const uint4*)(ap);
    a.q[1] = *(const uint4*)(ap + 8);
#pragma unroll
    for (int nb = 0; nb < 4; ++nb) {
        const unsigned short* bp = Bt + boff + (size_t)nb * 16 * Kst + k0;
        b[nb].q[0] = *(const uint4*)(bp);
        b[nb].q[1] = *(const uint4*)(bp + 8);
    }
}

// ---------------------------------------------------------------------------
// WMMA GEMM: Out[M,N](bf16) = relu(A[M,K](bf16) @ B[K,N] + bias[N])
//   Bt: bf16, stored transposed [N,K]. One wave computes a 16(M) x 64(N)
//   tile; K stepped by 32; double-buffered register fragments; LDS-staged
//   coalesced epilogue stores. Blocks are m0-major (8 waves share the A rows
//   -> streamed matrix read once from HBM; duplicate loads absorbed by WGP$).
// A-frag (16x32 bf16): lane -> M = l&15, half = l>>4,
//   VGPR r: K = 2r + 16*half
// B-frag (32x16 bf16): lane -> N = l&15, VGPR r: K = 2r + 16*half
// C/D 16x16 f32: VGPR r -> M = r + 8*half, N = l&15
// ---------------------------------------------------------------------------
__global__ __launch_bounds__(256) void wmma_gemm_relu(
    const unsigned short* __restrict__ A, const unsigned short* __restrict__ Bt,
    const float* __restrict__ bias, unsigned short* __restrict__ Out,
    int M, int N, int K) {

    __shared__ unsigned short stile[8][16 * 64];   // 2KB per wave

    const int lane = threadIdx.x & 31;
    const int wib  = threadIdx.x >> 5;
    const int tilesN = N >> 6;                     // 64-wide N tiles
    const int wave = blockIdx.x * 8 + wib;
    const int m0 = (wave / tilesN) << 4;
    const int n0 = (wave % tilesN) << 6;
    if (m0 >= M) return;

    const int half = lane >> 4;
    const int l16  = lane & 15;

    const size_t aoff = (size_t)(m0 + l16) * K + 16 * half;
    const size_t boff = (size_t)(n0 + l16) * K + 16 * half;

    v8f acc[4] = {};

    Frag a, b[4];
    load_frags(A, Bt, aoff, boff, K, 0, a, b);

    int k0 = 0;
    for (; k0 + 32 < K; k0 += 32) {
        Frag an, bn[4];
        load_frags(A, Bt, aoff, boff, K, k0 + 32, an, bn);
        if (k0 + 64 < K)   // hint the streamed A matrix (global_prefetch_b8)
            __builtin_prefetch(A + aoff + k0 + 64, 0, 1);
#pragma unroll
        for (int nb = 0; nb < 4; ++nb)
            acc[nb] = __builtin_amdgcn_wmma_f32_16x16x32_bf16(
                false, a.v, false, b[nb].v, (short)0, acc[nb], false, false);
        a = an;
#pragma unroll
        for (int nb = 0; nb < 4; ++nb) b[nb] = bn[nb];
    }
#pragma unroll
    for (int nb = 0; nb < 4; ++nb)
        acc[nb] = __builtin_amdgcn_wmma_f32_16x16x32_bf16(
            false, a.v, false, b[nb].v, (short)0, acc[nb], false, false);

    // Epilogue: bias + relu + bf16, staged through wave-private LDS so global
    // stores are coalesced b32 rows (same-wave DS ops are in order).
    unsigned short* tile = stile[wib];
#pragma unroll
    for (int nb = 0; nb < 4; ++nb) {
        const int col_l = nb * 16 + l16;
        const float bv = bias[n0 + col_l];
#pragma unroll
        for (int r = 0; r < 8; ++r) {
            const int row_l = r + 8 * half;
            float v = acc[nb][r] + bv;
            v = v > 0.f ? v : 0.f;
            tile[row_l * 64 + col_l] = f2bf(v);
        }
    }
    unsigned int* Out32 = (unsigned int*)Out;
    const unsigned int* t32 = (const unsigned int*)tile;
    const int rowStride = N >> 1;
#pragma unroll
    for (int r = 0; r < 16; ++r) {
        Out32[(size_t)(m0 + r) * rowStride + (n0 >> 1) + lane] = t32[r * 32 + lane];
    }
}

// ---------------------------------------------------------------------------
// out[m] = X[m,:512](bf16) . vec(f32) + bias[0]   (one wave per row)
// ---------------------------------------------------------------------------
__global__ void dot512(const unsigned short* __restrict__ X, const float* __restrict__ vec,
                       const float* __restrict__ bias, float* __restrict__ out, int M) {
    const int wave = (blockIdx.x * blockDim.x + threadIdx.x) >> 5;
    const int lane = threadIdx.x & 31;
    if (wave >= M) return;
    union { uint4 q[2]; unsigned int u[8]; } xr;
    const unsigned short* row = X + (size_t)wave * 512 + lane * 16;
    xr.q[0] = *(const uint4*)(row);
    xr.q[1] = *(const uint4*)(row + 8);
    const float* vp = vec + lane * 16;
    float s = 0.f;
#pragma unroll
    for (int t = 0; t < 8; ++t) {
        unsigned int u = xr.u[t];
        s += bf2f((unsigned short)(u & 0xffffu)) * vp[2 * t];
        s += bf2f((unsigned short)(u >> 16))     * vp[2 * t + 1];
    }
    s = wave_sum(s);
    if (lane == 0) out[wave] = s + bias[0];
}

// ---------------------------------------------------------------------------
// Per-span masked softmax over attn logits + weighted embed sum. Emits g rows
// twice: fp32 straight into d_out's g slice (required output) and bf16 into
// workspace (feeds the WMMA GEMM with zero in-loop conversion). One wave/span.
// ---------------------------------------------------------------------------
__global__ void span_gather(const float* __restrict__ states, const float* __restrict__ embeds,
                            const int* __restrict__ starts, const int* __restrict__ widths,
                            const float* __restrict__ attns, float* __restrict__ gout,
                            unsigned short* __restrict__ gbf, int n) {
    const int wave = (blockIdx.x * blockDim.x + threadIdx.x) >> 5;
    const int lane = threadIdx.x & 31;
    if (wave >= n) return;
    const int s = starts[wave];
    const int w = widths[wave];

    const bool valid = (lane <= w) && (lane < L_SPAN);
    float logit = valid ? attns[s + lane] : -__builtin_inff();
    float mx = wave_max(logit);
    float e = valid ? __expf(logit - mx) : 0.f;
    float denom = wave_sum(e);
    float wgt = e / denom;

    // each lane owns a contiguous 16-float chunk: d = lane*16 .. lane*16+15
    float acc[16];
#pragma unroll
    for (int t = 0; t < 16; ++t) acc[t] = 0.f;
    for (int j = 0; j <= w; ++j) {
        float wj = __shfl(wgt, j, 32);
        const float4* er = (const float4*)(embeds + (size_t)(s + j) * D_FEAT + lane * 16);
#pragma unroll
        for (int t = 0; t < 4; ++t) {
            float4 v = er[t];
            acc[4 * t + 0] += wj * v.x;
            acc[4 * t + 1] += wj * v.y;
            acc[4 * t + 2] += wj * v.z;
            acc[4 * t + 3] += wj * v.w;
        }
    }

    float* g = gout + (size_t)wave * G_DIM;
    unsigned short* gb = gbf + (size_t)wave * G_DIM;
    const float4* r0 = (const float4*)(states + (size_t)s * D_FEAT + lane * 16);
    const float4* r1 = (const float4*)(states + (size_t)(s + w) * D_FEAT + lane * 16);
    float4* g0 = (float4*)(g + lane * 16);
    float4* g1 = (float4*)(g + 512 + lane * 16);
    float4* g2 = (float4*)(g + 1024 + lane * 16);

    float4 v0[4], v1[4], v2[4];
#pragma unroll
    for (int t = 0; t < 4; ++t) {
        v0[t] = r0[t];
        v1[t] = r1[t];
        v2[t].x = acc[4 * t + 0]; v2[t].y = acc[4 * t + 1];
        v2[t].z = acc[4 * t + 2]; v2[t].w = acc[4 * t + 3];
        g0[t] = v0[t];
        g1[t] = v1[t];
        g2[t] = v2[t];
    }
    // bf16 copy (packed, 2x uint4 per 16-element segment chunk)
    union { unsigned int u[8]; uint4 q[2]; } p;
#pragma unroll
    for (int t = 0; t < 4; ++t) {
        p.u[2 * t]     = pack2bf(v0[t].x, v0[t].y);
        p.u[2 * t + 1] = pack2bf(v0[t].z, v0[t].w);
    }
    *(uint4*)(gb + lane * 16) = p.q[0];
    *(uint4*)(gb + lane * 16 + 8) = p.q[1];
#pragma unroll
    for (int t = 0; t < 4; ++t) {
        p.u[2 * t]     = pack2bf(v1[t].x, v1[t].y);
        p.u[2 * t + 1] = pack2bf(v1[t].z, v1[t].w);
    }
    *(uint4*)(gb + 512 + lane * 16) = p.q[0];
    *(uint4*)(gb + 512 + lane * 16 + 8) = p.q[1];
#pragma unroll
    for (int t = 0; t < 4; ++t) {
        p.u[2 * t]     = pack2bf(v2[t].x, v2[t].y);
        p.u[2 * t + 1] = pack2bf(v2[t].z, v2[t].w);
    }
    *(uint4*)(gb + 1024 + lane * 16) = p.q[0];
    *(uint4*)(gb + 1024 + lane * 16 + 8) = p.q[1];
}

// ---------------------------------------------------------------------------
// Top-k by rank counting (strict total order via index tie-break => ranks are
// a permutation => exactly kkeep elements have rank < kkeep).
// ---------------------------------------------------------------------------
__global__ void rank_topk(const float* __restrict__ scores, int* __restrict__ flags,
                          int n, int kkeep) {
    __shared__ float tile[256];
    const int i = blockIdx.x * 256 + threadIdx.x;
    const float si = (i < n) ? scores[i] : 0.f;
    int rank = 0;
    for (int base = 0; base < n; base += 256) {
        int j = base + threadIdx.x;
        tile[threadIdx.x] = (j < n) ? scores[j] : -__builtin_inff();
        __syncthreads();
#pragma unroll 8
        for (int t = 0; t < 256; ++t) {
            const int jj = base + t;
            const float sj = tile[t];
            rank += ((sj > si) || (sj == si && jj < i)) ? 1 : 0;
        }
        __syncthreads();
    }
    if (i < n) flags[i] = (rank < kkeep) ? 1 : 0;
}

__global__ void select_write(const float* __restrict__ scores, const int* __restrict__ flags,
                             float* __restrict__ out_scores, float* __restrict__ out_idx, int n) {
    const int i = blockIdx.x * 256 + threadIdx.x;
    if (i >= n || !flags[i]) return;
    int pos = 0;
    for (int j = 0; j < i; ++j) pos += flags[j];
    out_scores[pos] = scores[i];
    out_idx[pos]    = (float)i;   // output buffer is fp32; indices cast
}

// ---------------------------------------------------------------------------
extern "C" void kernel_launch(void* const* d_in, const int* in_sizes, int n_in,
                              void* d_out, int out_size, void* d_ws, size_t ws_size,
                              hipStream_t stream) {
    const float* states = (const float*)d_in[0];
    const float* embeds = (const float*)d_in[1];
    const int*   sstart = (const int*)d_in[2];
    const int*   swidth = (const int*)d_in[3];
    const float* W_a1 = (const float*)d_in[4];  const float* b_a1 = (const float*)d_in[5];
    const float* W_a2 = (const float*)d_in[6];  const float* b_a2 = (const float*)d_in[7];
    const float* w_a3 = (const float*)d_in[8];  const float* b_a3 = (const float*)d_in[9];
    const float* W_s1 = (const float*)d_in[10]; const float* b_s1 = (const float*)d_in[11];
    const float* W_s2 = (const float*)d_in[12]; const float* b_s2 = (const float*)d_in[13];
    const float* w_s3 = (const float*)d_in[14]; const float* b_s3 = (const float*)d_in[15];
    float* out = (float*)d_out;

    // d_out layout: [scores_pruned: K_PRUNE][g: N_SPAN*G_DIM][pruned_idx: K_PRUNE]
    float* g_out   = out + K_PRUNE;
    float* idx_out = out + K_PRUNE + (size_t)N_SPAN * G_DIM;

    // workspace carve-up
    char* ws = (char*)d_ws;
    auto carve = [&](size_t bytes) -> char* {
        char* p = ws;
        ws += (bytes + 255) & ~(size_t)255;
        return p;
    };
    unsigned short* Wa1t = (unsigned short*)carve((size_t)H_DIM * D_FEAT * 2);
    unsigned short* Wa2t = (unsigned short*)carve((size_t)H_DIM * H_DIM * 2);
    unsigned short* Ws1t = (unsigned short*)carve((size_t)H_DIM * G_DIM * 2);
    unsigned short* Ws2t = (unsigned short*)carve((size_t)H_DIM * H_DIM * 2);
    unsigned short* stbf = (unsigned short*)carve((size_t)T_TOK * D_FEAT * 2);
    unsigned short* h1   = (unsigned short*)carve((size_t)T_TOK * H_DIM * 2);
    unsigned short* h2   = (unsigned short*)carve((size_t)T_TOK * H_DIM * 2);
    float*          attn = (float*)carve((size_t)T_TOK * 4);
    unsigned short* gbf  = (unsigned short*)carve((size_t)N_SPAN * G_DIM * 2);
    unsigned short* s1   = (unsigned short*)carve((size_t)N_SPAN * H_DIM * 2);
    unsigned short* s2   = (unsigned short*)carve((size_t)N_SPAN * H_DIM * 2);
    float*          scr  = (float*)carve((size_t)N_SPAN * 4);
    int*            flg  = (int*)carve((size_t)N_SPAN * 4);

    // 1) weights -> transposed bf16; states -> bf16
    convert_wt<<<(D_FEAT * H_DIM + 255) / 256, 256, 0, stream>>>(W_a1, Wa1t, D_FEAT, H_DIM);
    convert_wt<<<(H_DIM * H_DIM + 255) / 256, 256, 0, stream>>>(W_a2, Wa2t, H_DIM, H_DIM);
    convert_wt<<<(G_DIM * H_DIM + 255) / 256, 256, 0, stream>>>(W_s1, Ws1t, G_DIM, H_DIM);
    convert_wt<<<(H_DIM * H_DIM + 255) / 256, 256, 0, stream>>>(W_s2, Ws2t, H_DIM, H_DIM);
    convert_vec<<<(T_TOK * D_FEAT / 8 + 255) / 256, 256, 0, stream>>>(states, stbf, T_TOK * D_FEAT / 8);

    // 2) attention MLP (WMMA, pure bf16 inner loops)
    {
        int blocks = (T_TOK / 16) * (H_DIM / 64) / 8;   // 256
        wmma_gemm_relu<<<blocks, 256, 0, stream>>>(stbf, Wa1t, b_a1, h1, T_TOK, H_DIM, D_FEAT);
        wmma_gemm_relu<<<blocks, 256, 0, stream>>>(h1, Wa2t, b_a2, h2, T_TOK, H_DIM, H_DIM);
        dot512<<<T_TOK / 8, 256, 0, stream>>>(h2, w_a3, b_a3, attn, T_TOK);
    }

    // 3) span softmax + weighted embeds -> g (fp32 to d_out, bf16 to ws)
    span_gather<<<N_SPAN / 8, 256, 0, stream>>>(states, embeds, sstart, swidth, attn,
                                                g_out, gbf, N_SPAN);

    // 4) mention-score MLP (WMMA, pure bf16 inner loops)
    {
        int blocks = (N_SPAN / 16) * (H_DIM / 64) / 8;  // 2560
        wmma_gemm_relu<<<blocks, 256, 0, stream>>>(gbf, Ws1t, b_s1, s1, N_SPAN, H_DIM, G_DIM);
        wmma_gemm_relu<<<blocks, 256, 0, stream>>>(s1, Ws2t, b_s2, s2, N_SPAN, H_DIM, H_DIM);
        dot512<<<N_SPAN / 8, 256, 0, stream>>>(s2, w_s3, b_s3, scr, N_SPAN);
    }

    // 5) top-k prune, index-sorted
    rank_topk<<<N_SPAN / 256, 256, 0, stream>>>(scr, flg, N_SPAN, K_PRUNE);
    select_write<<<N_SPAN / 256, 256, 0, stream>>>(scr, flg, out, idx_out, N_SPAN);
}
